// TMSA_49813030699702
// MI455X (gfx1250) — matrix operand order repaired and checked by reading, not compiled, not run on hardware
//
#include <hip/hip_runtime.h>
#include <hip/hip_bf16.h>
#include <stdint.h>
#include <stddef.h>

typedef __bf16 bf16;
typedef __attribute__((ext_vector_type(16))) __bf16 v16bf;
typedef __attribute__((ext_vector_type(8)))  __bf16 v8bf;
typedef __attribute__((ext_vector_type(8)))  float  f32x8;

#define SCALE_Q 0.2041241452319315f   /* 24^-0.5 */
#define TWO_PI  6.283185307179586f

__device__ __forceinline__ f32x8 wmma_bf16(v16bf a, v16bf b, f32x8 c) {
  return __builtin_amdgcn_wmma_f32_16x16x32_bf16(false, a, false, b, (short)0, c, false, false);
}

union frag_u { v16bf v; v8bf h[2]; };

// A operand: [16(M) x 32(K)] tile, row-major with stride ld.
// ISA layout: lanes<16 -> M=lane, K={0..7,16..23}; lanes>=16 -> M=lane-16, K={8..15,24..31}.
__device__ __forceinline__ v16bf load_frag_a(const bf16* p, int ld, int lane) {
  int row = lane & 15;
  int kb  = (lane >> 4) << 3;            // 0 or 8
  const bf16* q = p + row * ld + kb;
  frag_u u;
  u.h[0] = *(const v8bf*)(q);
  u.h[1] = *(const v8bf*)(q + 16);
  return u.v;
}

// B operand from B^T storage: [16(N) x 32(K)] tile, row-major stride ld.
// ISA layout: lanes<16 -> N=lane, K={0..7,16..23}; lanes>=16 -> N=lane-16, K={8..15,24..31}.
__device__ __forceinline__ v16bf load_frag_b(const bf16* p, int ld, int lane) {
  int n  = lane & 15;
  int kb = (lane >> 4) << 3;             // 0 or 8
  const bf16* q = p + n * ld + kb;
  frag_u u;
  u.h[0] = *(const v8bf*)(q);
  u.h[1] = *(const v8bf*)(q + 16);
  return u.v;
}

/* ------------------------------------------------------------------ */
__global__ void k_fill0(uint32_t* p, size_t n) {
  for (size_t i = (size_t)blockIdx.x * blockDim.x + threadIdx.x; i < n;
       i += (size_t)gridDim.x * blockDim.x)
    p[i] = 0u;
}

/* dst[n*K + k] = bf16(src[k*N + n]) */
__global__ void k_transpose(const float* __restrict__ src, bf16* __restrict__ dst,
                            int K, int N) {
  int total = K * N;
  for (int i = blockIdx.x * blockDim.x + threadIdx.x; i < total;
       i += gridDim.x * blockDim.x) {
    int k = i / N, n = i % N;
    dst[(size_t)n * K + k] = (bf16)src[i];
  }
}

/* ------------------------------------------------------------------ */
/* LN1 + roll(-1,-4,-4) + window partition; xm = xw + sine pos bias.   */
__global__ void k_ln1(const float* __restrict__ x, const float* __restrict__ nw,
                      const float* __restrict__ nb, bf16* __restrict__ xw,
                      bf16* __restrict__ xm) {
  int lane = threadIdx.x & 31, wave = threadIdx.x >> 5;
  int t = blockIdx.x * 8 + wave;                 // windowed token 0..49151
  int win = t >> 7, tok = t & 127;
  int b = win / 192, r = win % 192;
  int dd = r >> 6, hh = (r >> 3) & 7, ww = r & 7;
  int td = tok >> 6, th = (tok >> 3) & 7, tw = tok & 7;
  int d = dd * 2 + td, h = hh * 8 + th, w = ww * 8 + tw;
  int sd = (d + 1) % 6, sh = (h + 4) & 63, sw = (w + 4) & 63;
  const float* src = x + ((((size_t)b * 6 + sd) * 64 + sh) * 64 + sw) * 192;

  int c0 = lane * 6;
  float v[6]; float s = 0.f;
#pragma unroll
  for (int i = 0; i < 6; i++) { v[i] = src[c0 + i]; s += v[i]; }
#pragma unroll
  for (int m = 16; m >= 1; m >>= 1) s += __shfl_xor(s, m, 32);
  float mu = s * (1.f / 192.f);
  float var = 0.f;
#pragma unroll
  for (int i = 0; i < 6; i++) { float dv = v[i] - mu; var += dv * dv; }
#pragma unroll
  for (int m = 16; m >= 1; m >>= 1) var += __shfl_xor(var, m, 32);
  float rs = rsqrtf(var * (1.f / 192.f) + 1e-5f);

  int pos = tok & 63; int ph = pos >> 3, pw = pos & 7;
  float yv = (float)(ph + 1) / (8.f + 1e-6f) * TWO_PI;
  float xv = (float)(pw + 1) / (8.f + 1e-6f) * TWO_PI;

  size_t dst = (size_t)t * 192 + c0;
#pragma unroll
  for (int i = 0; i < 6; i++) {
    int c = c0 + i;
    float ln = (v[i] - mu) * rs * nw[c] + nb[c];
    xw[dst + i] = (bf16)ln;
    int cc = c < 96 ? c : c - 96;
    float base = c < 96 ? yv : xv;
    float dim_t = powf(10000.f, (float)(cc >> 1) / 48.f);
    float ang = base / dim_t;
    float pb = (cc & 1) ? cosf(ang) : sinf(ang);
    xm[dst + i] = (bf16)(ln + pb);
  }
}

/* ------------------------------------------------------------------ */
/* QKV GEMM: 32x64 per wave, double-buffered fragments.                */
/* A[49152x192] @ Wt[576x192]^T -> scatter Q (scaled), K, V^T (pad 32) */
__global__ __launch_bounds__(128) void k_gemm_qkv(
    const bf16* __restrict__ A, const bf16* __restrict__ BT,
    const float* __restrict__ bias, bf16* __restrict__ qb,
    bf16* __restrict__ kb, bf16* __restrict__ vtb) {
  int lane = threadIdx.x & 31, wave = threadIdx.x >> 5;
  int m0 = blockIdx.x * 128 + wave * 32;
  int n0 = blockIdx.y * 64;
  const bf16* Ab = A + (size_t)m0 * 192;
  const bf16* Bb = BT + (size_t)n0 * 192;

  f32x8 acc[2][4] = {};
  v16bf a0 = load_frag_a(Ab, 192, lane);
  v16bf a1 = load_frag_a(Ab + 16 * 192, 192, lane);
  v16bf b0 = load_frag_b(Bb, 192, lane);
  v16bf b1 = load_frag_b(Bb + 16 * 192, 192, lane);
  v16bf b2 = load_frag_b(Bb + 32 * 192, 192, lane);
  v16bf b3 = load_frag_b(Bb + 48 * 192, 192, lane);

#pragma unroll
  for (int ks = 0; ks < 6; ks++) {
    int kn = (ks < 5 ? ks + 1 : 5) * 32;
    v16bf na0 = load_frag_a(Ab + kn, 192, lane);
    v16bf na1 = load_frag_a(Ab + 16 * 192 + kn, 192, lane);
    v16bf nb0 = load_frag_b(Bb + kn, 192, lane);
    v16bf nb1 = load_frag_b(Bb + 16 * 192 + kn, 192, lane);
    v16bf nb2 = load_frag_b(Bb + 32 * 192 + kn, 192, lane);
    v16bf nb3 = load_frag_b(Bb + 48 * 192 + kn, 192, lane);
    acc[0][0] = wmma_bf16(a0, b0, acc[0][0]);
    acc[1][0] = wmma_bf16(a1, b0, acc[1][0]);
    acc[0][1] = wmma_bf16(a0, b1, acc[0][1]);
    acc[1][1] = wmma_bf16(a1, b1, acc[1][1]);
    acc[0][2] = wmma_bf16(a0, b2, acc[0][2]);
    acc[1][2] = wmma_bf16(a1, b2, acc[1][2]);
    acc[0][3] = wmma_bf16(a0, b3, acc[0][3]);
    acc[1][3] = wmma_bf16(a1, b3, acc[1][3]);
    a0 = na0; a1 = na1; b0 = nb0; b1 = nb1; b2 = nb2; b3 = nb3;
  }

#pragma unroll
  for (int mi = 0; mi < 2; mi++) {
    int mrow = m0 + mi * 16 + ((lane >> 4) << 3);
#pragma unroll
    for (int cf = 0; cf < 4; cf++) {
      int n = n0 + cf * 16 + (lane & 15);
      int qkv = n / 192, rem = n % 192;
      int head = rem / 24, hd = rem % 24;
      float bv = bias[n];
#pragma unroll
      for (int j = 0; j < 8; j++) {
        int m = mrow + j;
        int win = m >> 7, tok = m & 127;
        float val = acc[mi][cf][j] + bv;
        size_t hb = (size_t)win * 8 + head;
        if (qkv == 0)      qb [(hb * 128 + tok) * 32 + hd] = (bf16)(val * SCALE_Q);
        else if (qkv == 1) kb [(hb * 128 + tok) * 32 + hd] = (bf16)val;
        else               vtb[(hb * 32 + hd) * 128 + tok] = (bf16)val;
      }
    }
  }
}

/* ------------------------------------------------------------------ */
/* Self-attention: one block per (window, head); 8 waves x 16 q-rows.  */
__global__ __launch_bounds__(256) void k_attn_self(
    const bf16* __restrict__ qb, const bf16* __restrict__ kb,
    const bf16* __restrict__ vtb, const float* __restrict__ rpb,
    const float* __restrict__ mask, bf16* __restrict__ xout) {
  __shared__ __align__(16) bf16 probs[8][16][128];   // 32 KB
  int lane = threadIdx.x & 31, wave = threadIdx.x >> 5;
  int win = blockIdx.x, head = blockIdx.y;
  int wloc = win % 192;
  size_t hb = (size_t)win * 8 + head;
  const bf16* Q  = qb  + hb * 128 * 32;
  const bf16* K  = kb  + hb * 128 * 32;
  const bf16* VT = vtb + hb * 32 * 128;
  int q0 = wave * 16;
  int ncol = lane & 15, mhi = (lane >> 4) << 3;

  f32x8 acc[8] = {};
  v16bf a = load_frag_a(Q + q0 * 32, 32, lane);
#pragma unroll
  for (int nf = 0; nf < 8; nf++) {
    v16bf bb = load_frag_b(K + nf * 16 * 32, 32, lane);
    acc[nf] = wmma_bf16(a, bb, acc[nf]);
  }
  /* relative position bias + shift mask */
#pragma unroll
  for (int nf = 0; nf < 8; nf++) {
    int jc = nf * 16 + ncol;
    int dj = jc >> 6, hj = (jc >> 3) & 7, wj = jc & 7;
#pragma unroll
    for (int j = 0; j < 8; j++) {
      int ir = q0 + mhi + j;
      int di = ir >> 6, hi = (ir >> 3) & 7, wi = ir & 7;
      int idx = (di - dj + 1) * 225 + (hi - hj + 7) * 15 + (wi - wj + 7);
      acc[nf][j] += rpb[idx * 8 + head] +
                    mask[((size_t)wloc * 128 + ir) * 128 + jc];
    }
  }
  /* softmax over 128 cols (8 frags x 16-lane half) */
#pragma unroll
  for (int j = 0; j < 8; j++) {
    float mx = -3.4e38f;
#pragma unroll
    for (int nf = 0; nf < 8; nf++) mx = fmaxf(mx, acc[nf][j]);
#pragma unroll
    for (int m = 8; m >= 1; m >>= 1) mx = fmaxf(mx, __shfl_xor(mx, m, 32));
    float sum = 0.f;
#pragma unroll
    for (int nf = 0; nf < 8; nf++) { float e = expf(acc[nf][j] - mx); acc[nf][j] = e; sum += e; }
#pragma unroll
    for (int m = 8; m >= 1; m >>= 1) sum += __shfl_xor(sum, m, 32);
    float inv = 1.f / sum;
    int row = mhi + j;
#pragma unroll
    for (int nf = 0; nf < 8; nf++)
      probs[wave][row][nf * 16 + ncol] = (bf16)(acc[nf][j] * inv);
  }
  /* AV: K=128 in 4 steps, N = padded HD (2 tiles) */
  f32x8 o[2] = {};
#pragma unroll
  for (int ks = 0; ks < 4; ks++) {
    v16bf pa = load_frag_a(&probs[wave][0][ks * 32], 128, lane);
#pragma unroll
    for (int nt = 0; nt < 2; nt++) {
      v16bf vb = load_frag_b(VT + nt * 16 * 128 + ks * 32, 128, lane);
      o[nt] = wmma_bf16(pa, vb, o[nt]);
    }
  }
#pragma unroll
  for (int nt = 0; nt < 2; nt++) {
    int hd = nt * 16 + ncol;
    if (hd < 24) {
#pragma unroll
      for (int j = 0; j < 8; j++) {
        int tok = q0 + mhi + j;
        xout[((size_t)win * 128 + tok) * 384 + 192 + head * 24 + hd] = (bf16)o[nt][j];
      }
    }
  }
}

/* ------------------------------------------------------------------ */
/* Mutual attention: n=64; blockIdx.z chooses (q2,k1,v1) vs (q1,k2,v2) */
__global__ __launch_bounds__(128) void k_attn_mut(
    const bf16* __restrict__ qb, const bf16* __restrict__ kb,
    const bf16* __restrict__ vtb, const float* __restrict__ mask,
    bf16* __restrict__ xout) {
  __shared__ __align__(16) bf16 probs[4][16][64];    // 8 KB
  int lane = threadIdx.x & 31, wave = threadIdx.x >> 5;
  int win = blockIdx.x, head = blockIdx.y, which = blockIdx.z;
  int wloc = win % 192;
  int qoff = which == 0 ? 64 : 0;
  int koff = which == 0 ? 0 : 64;
  size_t hb = (size_t)win * 8 + head;
  const bf16* Q  = qb  + hb * 128 * 32;
  const bf16* K  = kb  + hb * 128 * 32;
  const bf16* VT = vtb + hb * 32 * 128;
  int q0 = wave * 16;
  int ncol = lane & 15, mhi = (lane >> 4) << 3;

  f32x8 acc[4] = {};
  v16bf a = load_frag_a(Q + (qoff + q0) * 32, 32, lane);
#pragma unroll
  for (int nf = 0; nf < 4; nf++) {
    v16bf bb = load_frag_b(K + (koff + nf * 16) * 32, 32, lane);
    acc[nf] = wmma_bf16(a, bb, acc[nf]);
  }
#pragma unroll
  for (int nf = 0; nf < 4; nf++) {
    int jc = nf * 16 + ncol;
#pragma unroll
    for (int j = 0; j < 8; j++) {
      int ir = q0 + mhi + j;                      // row within this attn call
      acc[nf][j] += mask[((size_t)wloc * 128 + ir) * 128 + jc];
    }
  }
#pragma unroll
  for (int j = 0; j < 8; j++) {
    float mx = -3.4e38f;
#pragma unroll
    for (int nf = 0; nf < 4; nf++) mx = fmaxf(mx, acc[nf][j]);
#pragma unroll
    for (int m = 8; m >= 1; m >>= 1) mx = fmaxf(mx, __shfl_xor(mx, m, 32));
    float sum = 0.f;
#pragma unroll
    for (int nf = 0; nf < 4; nf++) { float e = expf(acc[nf][j] - mx); acc[nf][j] = e; sum += e; }
#pragma unroll
    for (int m = 8; m >= 1; m >>= 1) sum += __shfl_xor(sum, m, 32);
    float inv = 1.f / sum;
    int row = mhi + j;
#pragma unroll
    for (int nf = 0; nf < 4; nf++)
      probs[wave][row][nf * 16 + ncol] = (bf16)(acc[nf][j] * inv);
  }
  f32x8 o[2] = {};
#pragma unroll
  for (int ks = 0; ks < 2; ks++) {
    v16bf pa = load_frag_a(&probs[wave][0][ks * 32], 64, lane);
#pragma unroll
    for (int nt = 0; nt < 2; nt++) {
      v16bf vb = load_frag_b(VT + nt * 16 * 128 + koff + ks * 32, 128, lane);
      o[nt] = wmma_bf16(pa, vb, o[nt]);
    }
  }
#pragma unroll
  for (int nt = 0; nt < 2; nt++) {
    int hd = nt * 16 + ncol;
    if (hd < 24) {
#pragma unroll
      for (int j = 0; j < 8; j++) {
        int tok = which * 64 + q0 + mhi + j;      // concat([x1,x2], token dim)
        xout[((size_t)win * 128 + tok) * 384 + head * 24 + hd] = (bf16)o[nt][j];
      }
    }
  }
}

/* ------------------------------------------------------------------ */
/* proj GEMM (32x64/wave, double-buffered) + window reverse + residual */
__global__ __launch_bounds__(128) void k_gemm_proj(
    const bf16* __restrict__ A, const bf16* __restrict__ BT,
    const float* __restrict__ bias, const float* __restrict__ xin,
    float* __restrict__ out) {
  int lane = threadIdx.x & 31, wave = threadIdx.x >> 5;
  int m0 = blockIdx.x * 128 + wave * 32;
  int n0 = blockIdx.y * 64;
  const bf16* Ab = A + (size_t)m0 * 384;
  const bf16* Bb = BT + (size_t)n0 * 384;

  f32x8 acc[2][4] = {};
  v16bf a0 = load_frag_a(Ab, 384, lane);
  v16bf a1 = load_frag_a(Ab + 16 * 384, 384, lane);
  v16bf b0 = load_frag_b(Bb, 384, lane);
  v16bf b1 = load_frag_b(Bb + 16 * 384, 384, lane);
  v16bf b2 = load_frag_b(Bb + 32 * 384, 384, lane);
  v16bf b3 = load_frag_b(Bb + 48 * 384, 384, lane);

#pragma unroll
  for (int ks = 0; ks < 12; ks++) {
    int kn = (ks < 11 ? ks + 1 : 11) * 32;
    v16bf na0 = load_frag_a(Ab + kn, 384, lane);
    v16bf na1 = load_frag_a(Ab + 16 * 384 + kn, 384, lane);
    v16bf nb0 = load_frag_b(Bb + kn, 384, lane);
    v16bf nb1 = load_frag_b(Bb + 16 * 384 + kn, 384, lane);
    v16bf nb2 = load_frag_b(Bb + 32 * 384 + kn, 384, lane);
    v16bf nb3 = load_frag_b(Bb + 48 * 384 + kn, 384, lane);
    acc[0][0] = wmma_bf16(a0, b0, acc[0][0]);
    acc[1][0] = wmma_bf16(a1, b0, acc[1][0]);
    acc[0][1] = wmma_bf16(a0, b1, acc[0][1]);
    acc[1][1] = wmma_bf16(a1, b1, acc[1][1]);
    acc[0][2] = wmma_bf16(a0, b2, acc[0][2]);
    acc[1][2] = wmma_bf16(a1, b2, acc[1][2]);
    acc[0][3] = wmma_bf16(a0, b3, acc[0][3]);
    acc[1][3] = wmma_bf16(a1, b3, acc[1][3]);
    a0 = na0; a1 = na1; b0 = nb0; b1 = nb1; b2 = nb2; b3 = nb3;
  }

#pragma unroll
  for (int mi = 0; mi < 2; mi++) {
    int mrow = m0 + mi * 16 + ((lane >> 4) << 3);
#pragma unroll
    for (int cf = 0; cf < 4; cf++) {
      int n = n0 + cf * 16 + (lane & 15);
      float bv = bias[n];
#pragma unroll
      for (int j = 0; j < 8; j++) {
        int m = mrow + j;
        int win = m >> 7, tok = m & 127;
        int b = win / 192, r = win % 192;
        int dd = r >> 6, hh = (r >> 3) & 7, ww = r & 7;
        int d = dd * 2 + (tok >> 6), h = hh * 8 + ((tok >> 3) & 7), w = ww * 8 + (tok & 7);
        int dp = (d + 1) % 6, hp = (h + 4) & 63, wp = (w + 4) & 63;
        size_t oi = ((((size_t)b * 6 + dp) * 64 + hp) * 64 + wp) * 192 + n;
        out[oi] = xin[oi] + acc[mi][cf][j] + bv;
      }
    }
  }
}

/* ------------------------------------------------------------------ */
/* LN2 + gate logits + top-2 softmax -> xn bf16, dense cw[49152x8]     */
__global__ void k_ln2_gate(const float* __restrict__ xres,
                           const float* __restrict__ nw, const float* __restrict__ nb,
                           const float* __restrict__ gw, const float* __restrict__ gb,
                           bf16* __restrict__ xn, float* __restrict__ cw) {
  int lane = threadIdx.x & 31, wave = threadIdx.x >> 5;
  int t = blockIdx.x * 8 + wave;
  const float* src = xres + (size_t)t * 192;
  int c0 = lane * 6;
  float v[6]; float s = 0.f;
#pragma unroll
  for (int i = 0; i < 6; i++) { v[i] = src[c0 + i]; s += v[i]; }
#pragma unroll
  for (int m = 16; m >= 1; m >>= 1) s += __shfl_xor(s, m, 32);
  float mu = s * (1.f / 192.f);
  float var = 0.f;
#pragma unroll
  for (int i = 0; i < 6; i++) { float dv = v[i] - mu; var += dv * dv; }
#pragma unroll
  for (int m = 16; m >= 1; m >>= 1) var += __shfl_xor(var, m, 32);
  float rs = rsqrtf(var * (1.f / 192.f) + 1e-5f);

  float part[8] = {0,0,0,0,0,0,0,0};
#pragma unroll
  for (int i = 0; i < 6; i++) {
    int c = c0 + i;
    float ln = (v[i] - mu) * rs * nw[c] + nb[c];
    xn[(size_t)t * 192 + c] = (bf16)ln;
#pragma unroll
    for (int e = 0; e < 8; e++) part[e] += ln * gw[c * 8 + e];
  }
#pragma unroll
  for (int e = 0; e < 8; e++)
#pragma unroll
    for (int m = 16; m >= 1; m >>= 1) part[e] += __shfl_xor(part[e], m, 32);

  if (lane == 0) {
    float lg[8];
#pragma unroll
    for (int e = 0; e < 8; e++) lg[e] = part[e] + gb[e];
    int b0 = 0;
#pragma unroll
    for (int e = 1; e < 8; e++) if (lg[e] > lg[b0]) b0 = e;
    int b1 = (b0 == 0) ? 1 : 0;
#pragma unroll
    for (int e = 0; e < 8; e++) if (e != b0 && lg[e] > lg[b1]) b1 = e;
    float e1 = expf(lg[b1] - lg[b0]);
    float w0 = 1.f / (1.f + e1), w1 = e1 / (1.f + e1);
    float* o = cw + (size_t)t * 8;
#pragma unroll
    for (int e = 0; e < 8; e++) o[e] = 0.f;
    o[b0] = w0; o[b1] = w1;
  }
}

/* ------------------------------------------------------------------ */
/* Fused MoE expert: GEMM1+GELU (LDS h) then GEMM2, weighted += d_out  */
__global__ __launch_bounds__(128) void k_moe(
    const bf16* __restrict__ xn, const bf16* __restrict__ w1t,
    const float* __restrict__ b1, const bf16* __restrict__ w2t,
    const float* __restrict__ b2, const float* __restrict__ cw, int e,
    float* __restrict__ out) {
  __shared__ __align__(16) bf16 hbuf[64 * 384];      // 48 KB
  int lane = threadIdx.x & 31, wave = threadIdx.x >> 5;
  int m0 = blockIdx.x * 64 + wave * 16;
  bf16* hrow = &hbuf[wave * 16 * 384];
  int ncol = lane & 15, mhi = (lane >> 4) << 3;

  v16bf a6[6];
#pragma unroll
  for (int ks = 0; ks < 6; ks++)
    a6[ks] = load_frag_a(xn + (size_t)m0 * 192 + ks * 32, 192, lane);

  /* stage 1: h = gelu(xn @ W1 + b1), staged to LDS as bf16            */
  for (int nf = 0; nf < 24; nf++) {
    if (nf < 23)
      __builtin_prefetch(w1t + (size_t)((nf + 1) * 16) * 192 + lane * 96, 0, 1);
    v16bf bb[6];
#pragma unroll
    for (int ks = 0; ks < 6; ks++)
      bb[ks] = load_frag_b(w1t + (size_t)(nf * 16) * 192 + ks * 32, 192, lane);
    f32x8 acc = {};
#pragma unroll
    for (int ks = 0; ks < 6; ks++) acc = wmma_bf16(a6[ks], bb[ks], acc);
    int n = nf * 16 + ncol;
    float bv = b1[n];
#pragma unroll
    for (int j = 0; j < 8; j++) {
      float v = acc[j] + bv;
      float g = 0.5f * v * (1.f + erff(v * 0.7071067811865475f));
      hrow[(mhi + j) * 384 + n] = (bf16)g;
    }
  }

  /* stage 2: out += cw * (h @ W2 + b2); 4 accumulators per A-frag read */
#pragma unroll
  for (int nb4 = 0; nb4 < 3; nb4++) {
    f32x8 acc[4] = {};
#pragma unroll
    for (int ks = 0; ks < 12; ks++) {
      v16bf pa = load_frag_a(hrow + ks * 32, 384, lane);
#pragma unroll
      for (int f = 0; f < 4; f++) {
        v16bf bb = load_frag_b(w2t + (size_t)((nb4 * 4 + f) * 16) * 384 + ks * 32, 384, lane);
        acc[f] = wmma_bf16(pa, bb, acc[f]);
      }
    }
#pragma unroll
    for (int f = 0; f < 4; f++) {
      int n = (nb4 * 4 + f) * 16 + ncol;
      float bv = b2[n];
#pragma unroll
      for (int j = 0; j < 8; j++) {
        int m = m0 + mhi + j;
        float wgt = cw[(size_t)m * 8 + e];
        out[(size_t)m * 192 + n] += wgt * (acc[f][j] + bv);
      }
    }
  }
}

/* ================================================================== */
extern "C" void kernel_launch(void* const* d_in, const int* in_sizes, int n_in,
                              void* d_out, int out_size, void* d_ws, size_t ws_size,
                              hipStream_t stream) {
  (void)in_sizes; (void)n_in; (void)out_size; (void)ws_size;
  const float* x      = (const float*)d_in[0];
  const float* mask   = (const float*)d_in[1];
  const float* n1w    = (const float*)d_in[2];
  const float* n1b    = (const float*)d_in[3];
  const float* qkvsw  = (const float*)d_in[4];
  const float* qkvsb  = (const float*)d_in[5];
  const float* qkvmw  = (const float*)d_in[6];
  const float* qkvmb  = (const float*)d_in[7];
  const float* rpb    = (const float*)d_in[8];
  const float* projw  = (const float*)d_in[9];
  const float* projb  = (const float*)d_in[10];
  const float* n2w    = (const float*)d_in[11];
  const float* n2b    = (const float*)d_in[12];
  const float* gatew  = (const float*)d_in[13];
  const float* gateb  = (const float*)d_in[14];
  const float* W1     = (const float*)d_in[15];
  const float* b1     = (const float*)d_in[16];
  const float* W2     = (const float*)d_in[17];
  const float* b2     = (const float*)d_in[18];
  float* out = (float*)d_out;

  char* ws = (char*)d_ws;
  size_t off = 0;
  auto take = [&](size_t bytes) -> char* {
    char* r = ws + off;
    off += (bytes + 255) & ~(size_t)255;
    return r;
  };
  const size_t T = 49152;                    // total windowed tokens
  bf16* xw    = (bf16*)take(T * 192 * 2);
  bf16* xm    = (bf16*)take(T * 192 * 2);
  bf16* wqsT  = (bf16*)take((size_t)576 * 192 * 2);
  bf16* wqmT  = (bf16*)take((size_t)576 * 192 * 2);
  bf16* wprT  = (bf16*)take((size_t)192 * 384 * 2);
  bf16* w1T   = (bf16*)take((size_t)8 * 384 * 192 * 2);
  bf16* w2T   = (bf16*)take((size_t)8 * 192 * 384 * 2);
  bf16* qb    = (bf16*)take((size_t)384 * 8 * 128 * 32 * 2);
  bf16* kbuf  = (bf16*)take((size_t)384 * 8 * 128 * 32 * 2);
  bf16* vtb   = (bf16*)take((size_t)384 * 8 * 32 * 128 * 2);
  bf16* xout  = (bf16*)take(T * 384 * 2);
  bf16* xn    = (bf16*)take(T * 192 * 2);
  float* cw   = (float*)take(T * 8 * 4);

  /* zero Q/K/Vt (so HD pad lanes 24..31 are zero for the WMMA K dim) */
  size_t zwords = (3 * (size_t)384 * 8 * 128 * 32 * 2) / 4;
  k_fill0<<<4096, 256, 0, stream>>>((uint32_t*)qb, zwords);

  /* transpose+convert weights to bf16 [N x K] */
  k_transpose<<<432, 256, 0, stream>>>(qkvsw, wqsT, 192, 576);
  k_transpose<<<432, 256, 0, stream>>>(qkvmw, wqmT, 192, 576);
  k_transpose<<<288, 256, 0, stream>>>(projw, wprT, 384, 192);
  for (int e = 0; e < 8; e++) {
    k_transpose<<<288, 256, 0, stream>>>(W1 + (size_t)e * 192 * 384,
                                         w1T + (size_t)e * 384 * 192, 192, 384);
    k_transpose<<<288, 256, 0, stream>>>(W2 + (size_t)e * 384 * 192,
                                         w2T + (size_t)e * 192 * 384, 384, 192);
  }

  k_ln1<<<6144, 256, 0, stream>>>(x, n1w, n1b, xw, xm);

  /* self attention */
  k_gemm_qkv<<<dim3(384, 9), 128, 0, stream>>>(xw, wqsT, qkvsb, qb, kbuf, vtb);
  k_attn_self<<<dim3(384, 8), 256, 0, stream>>>(qb, kbuf, vtb, rpb, mask, xout);

  /* mutual attention (reuses q/k/vt buffers; pads stay zero) */
  k_gemm_qkv<<<dim3(384, 9), 128, 0, stream>>>(xm, wqmT, qkvmb, qb, kbuf, vtb);
  k_attn_mut<<<dim3(384, 8, 2), 128, 0, stream>>>(qb, kbuf, vtb, mask, xout);

  /* proj + window reverse + residual */
  k_gemm_proj<<<dim3(384, 3), 128, 0, stream>>>(xout, wprT, projb, x, out);

  /* LN2 + gate */
  k_ln2_gate<<<6144, 256, 0, stream>>>(out, n2w, n2b, gatew, gateb, xn, cw);

  /* MoE experts (sequential, deterministic accumulate into d_out) */
  for (int e = 0; e < 8; e++)
    k_moe<<<768, 128, 0, stream>>>(xn, w1T + (size_t)e * 384 * 192,
                                   b1 + (size_t)e * 384,
                                   w2T + (size_t)e * 192 * 384,
                                   b2 + (size_t)e * 192, cw, e, out);
}
